// AttentionBlock_25855703122610
// MI455X (gfx1250) — matrix operand order repaired and checked by reading, not compiled, not run on hardware
//
#include <hip/hip_runtime.h>
#include <hip/hip_bf16.h>

typedef __attribute__((ext_vector_type(16))) _Float16 v16h;
typedef __attribute__((ext_vector_type(8)))  _Float16 v8h;
typedef __attribute__((ext_vector_type(8)))  float    v8f;
typedef __attribute__((ext_vector_type(4)))  float    v4f;

#define B_   2
#define S_   2048
#define DM   2048
#define NH   16
#define NKV  4
#define HD   128
#define REP  (NH / NKV)          // 4
#define MROWS (B_ * S_)          // 4096
#define LN10K 9.210340371976184f // ln(10000)
#define QSCALE 0.08838834764831845f // 1/sqrt(128)

// ---------------------------------------------------------------------------
// Fragment loaders (match CDNA5 ISA 16-bit layouts)
// ---------------------------------------------------------------------------
// A 16x32 f16: lane t holds row (t&15); K = {h*8..h*8+7, 16+h*8..16+h*8+7}, h=t>>4
__device__ inline v16h load_afrag(const _Float16* __restrict__ base, int lda, int lane) {
  int r = lane & 15, h = lane >> 4;
  const _Float16* p = base + (size_t)r * lda;
  v16h f;
  *((v8h*)&f)     = *(const v8h*)(p + h * 8);
  *((v8h*)&f + 1) = *(const v8h*)(p + 16 + h * 8);
  return f;
}
// B 32x16 f16 from "BT" storage (row n = column n's K values, stride ldb):
// lane t holds column (t&15), K = {h*8..h*8+7, 16+h*8..16+h*8+7}
__device__ inline v16h load_bfrag(const _Float16* __restrict__ baseT, int ldb, int lane) {
  int n = lane & 15, h = lane >> 4;
  const _Float16* p = baseT + (size_t)n * ldb + h * 8;
  v16h f;
  *((v8h*)&f)     = *(const v8h*)(p);
  *((v8h*)&f + 1) = *(const v8h*)(p + 16);
  return f;
}

__device__ inline v8f wmma16(v16h a, v16h b, v8f c) {
  return __builtin_amdgcn_wmma_f32_16x16x32_f16(false, a, false, b, (short)0, c,
                                                false, false);
}

// Wave computes a 16(m) x 128(n) tile of A[M,K] * BT[N,K]^T, K-loop of 32,
// software-pipelined with a 2-stage double buffer. sched_barrier(0) pins the
// load groups so the 18 loads of stage N+1 stay in flight across the 8 WMMAs
// of stage N (instead of being sunk to their uses and serialized).
__device__ inline void wave_gemm_tile(const _Float16* __restrict__ A, int lda,
                                      const _Float16* __restrict__ BT, int ldb,
                                      int K, int lane, v8f acc[8]) {
#pragma unroll
  for (int nt = 0; nt < 8; ++nt) acc[nt] = v8f{};
  v16h b0[8], b1[8], a0, a1;
#pragma unroll
  for (int nt = 0; nt < 8; ++nt)
    b0[nt] = load_bfrag(BT + (size_t)(nt * 16) * ldb, ldb, lane);
  a0 = load_afrag(A, lda, lane);
  __builtin_amdgcn_sched_barrier(0);
  for (int k0 = 0; k0 < K; k0 += 64) {   // K must be a multiple of 64
    // prefetch stage k0+32
#pragma unroll
    for (int nt = 0; nt < 8; ++nt)
      b1[nt] = load_bfrag(BT + (size_t)(nt * 16) * ldb + k0 + 32, ldb, lane);
    a1 = load_afrag(A + k0 + 32, lda, lane);
    __builtin_amdgcn_sched_barrier(0);
    // compute stage k0
#pragma unroll
    for (int nt = 0; nt < 8; ++nt)
      acc[nt] = wmma16(a0, b0[nt], acc[nt]);
    __builtin_amdgcn_sched_barrier(0);
    // prefetch stage k0+64
    if (k0 + 64 < K) {
#pragma unroll
      for (int nt = 0; nt < 8; ++nt)
        b0[nt] = load_bfrag(BT + (size_t)(nt * 16) * ldb + k0 + 64, ldb, lane);
      a0 = load_afrag(A + k0 + 64, lda, lane);
    }
    __builtin_amdgcn_sched_barrier(0);
    // compute stage k0+32
#pragma unroll
    for (int nt = 0; nt < 8; ++nt)
      acc[nt] = wmma16(a1, b1[nt], acc[nt]);
    __builtin_amdgcn_sched_barrier(0);
  }
}

// ---------------------------------------------------------------------------
// Pack kernels
// ---------------------------------------------------------------------------
__global__ void cvt_x_kernel(const float* __restrict__ x, _Float16* __restrict__ xh, int n) {
  int i = blockIdx.x * blockDim.x + threadIdx.x;
  if (i < n) xh[i] = (_Float16)x[i];
}
// wq [DM][NH*HD] -> WqT [n][d]
__global__ void pack_wq_kernel(const float* __restrict__ wq, _Float16* __restrict__ wqT) {
  int i = blockIdx.x * blockDim.x + threadIdx.x;
  int n = i / DM, d = i % DM;
  wqT[i] = (_Float16)wq[(size_t)d * DM + n];
}
// wkv [DM][NKV*2*HD] -> WkvT: rows 0..511 = K cols, rows 512..1023 = V cols
__global__ void pack_wkv_kernel(const float* __restrict__ wkv, _Float16* __restrict__ wkvT) {
  int i = blockIdx.x * blockDim.x + threadIdx.x;
  int n = i / DM, d = i % DM;
  int src;
  if (n < NKV * HD) { src = (n / HD) * (2 * HD) + (n % HD); }
  else { int nn = n - NKV * HD; src = (nn / HD) * (2 * HD) + HD + (nn % HD); }
  wkvT[i] = (_Float16)wkv[(size_t)d * (NKV * 2 * HD) + src];
}
// wo [NH*HD][DM] -> WoT [o][hd]
__global__ void pack_wo_kernel(const float* __restrict__ wo, _Float16* __restrict__ woT) {
  int i = blockIdx.x * blockDim.x + threadIdx.x;
  int n = i / DM, k = i % DM;
  woT[i] = (_Float16)wo[(size_t)k * DM + n];
}

// ---------------------------------------------------------------------------
// Q projection + RoPE, writes Qh [B,NH,S,HD] f16 (pre-scaled by 1/sqrt(HD))
// ---------------------------------------------------------------------------
__global__ void __launch_bounds__(128)
qproj_rope_kernel(const _Float16* __restrict__ Xh, const _Float16* __restrict__ WqT,
                  const float* __restrict__ bq, const int* __restrict__ seq_pos,
                  _Float16* __restrict__ Qh) {
  int lane = threadIdx.x & 31, wave = threadIdx.x >> 5;
  int rowBase = (blockIdx.x * 4 + wave) * 16;
  int nBase = blockIdx.y * 128;           // one full head per wave tile
  v8f acc[8];
  wave_gemm_tile(Xh + (size_t)rowBase * DM, DM, WqT + (size_t)nBase * DM, DM, DM, lane, acc);
  int half = lane >> 4, nl = lane & 15;
#pragma unroll
  for (int nt = 0; nt < 8; ++nt) {
    float bb = bq[nBase + nt * 16 + nl];
#pragma unroll
    for (int v = 0; v < 8; ++v) acc[nt][v] += bb;
  }
  int h = nBase / HD;
  int b = rowBase / S_, s0 = rowBase % S_;
  int pos[8];
#pragma unroll
  for (int v = 0; v < 8; ++v) pos[v] = seq_pos[b * S_ + s0 + v + 8 * half];
#pragma unroll
  for (int nt = 0; nt < 4; ++nt) {
    int p = nt * 16 + nl;                 // rotation pair index 0..63
    float inv_ts = __expf(-(float)p * (LN10K / 64.f));
#pragma unroll
    for (int v = 0; v < 8; ++v) {
      float sn, cs;
      __sincosf((float)pos[v] * inv_ts, &sn, &cs);
      float x1 = acc[nt][v], x2 = acc[nt + 4][v];
      int s = s0 + v + 8 * half;
      size_t o = ((size_t)(b * NH + h) * S_ + s) * HD;
      Qh[o + p]      = (_Float16)((x1 * cs - x2 * sn) * QSCALE);
      Qh[o + p + 64] = (_Float16)((x2 * cs + x1 * sn) * QSCALE);
    }
  }
}

// ---------------------------------------------------------------------------
// KV projection: K half gets RoPE -> Kh [B,NKV,S,HD]; V half -> Vt [B,NKV,HD,S]
// ---------------------------------------------------------------------------
__global__ void __launch_bounds__(128)
kvproj_kernel(const _Float16* __restrict__ Xh, const _Float16* __restrict__ WkvT,
              const float* __restrict__ bkv, const int* __restrict__ seq_pos,
              _Float16* __restrict__ Kh, _Float16* __restrict__ Vt) {
  int lane = threadIdx.x & 31, wave = threadIdx.x >> 5;
  int rowBase = (blockIdx.x * 4 + wave) * 16;
  int nBase = blockIdx.y * 128;           // cols 0..511 K, 512..1023 V
  v8f acc[8];
  wave_gemm_tile(Xh + (size_t)rowBase * DM, DM, WkvT + (size_t)nBase * DM, DM, DM, lane, acc);
  int half = lane >> 4, nl = lane & 15;
  bool isK = nBase < NKV * HD;
  int kvh = (nBase % (NKV * HD)) / HD;
#pragma unroll
  for (int nt = 0; nt < 8; ++nt) {
    int n = nBase + nt * 16 + nl;
    int bi = isK ? ((n / HD) * 2 * HD + (n % HD))
                 : (((n - NKV * HD) / HD) * 2 * HD + HD + ((n - NKV * HD) % HD));
    float bb = bkv[bi];
#pragma unroll
    for (int v = 0; v < 8; ++v) acc[nt][v] += bb;
  }
  int b = rowBase / S_, s0 = rowBase % S_;
  if (isK) {
    int pos[8];
#pragma unroll
    for (int v = 0; v < 8; ++v) pos[v] = seq_pos[b * S_ + s0 + v + 8 * half];
#pragma unroll
    for (int nt = 0; nt < 4; ++nt) {
      int p = nt * 16 + nl;
      float inv_ts = __expf(-(float)p * (LN10K / 64.f));
#pragma unroll
      for (int v = 0; v < 8; ++v) {
        float sn, cs;
        __sincosf((float)pos[v] * inv_ts, &sn, &cs);
        float x1 = acc[nt][v], x2 = acc[nt + 4][v];
        int s = s0 + v + 8 * half;
        size_t o = ((size_t)(b * NKV + kvh) * S_ + s) * HD;
        Kh[o + p]      = (_Float16)(x1 * cs - x2 * sn);
        Kh[o + p + 64] = (_Float16)(x2 * cs + x1 * sn);
      }
    }
  } else {
#pragma unroll
    for (int nt = 0; nt < 8; ++nt) {
      int d = nt * 16 + nl;
#pragma unroll
      for (int v = 0; v < 8; ++v) {
        int s = s0 + v + 8 * half;
        Vt[((size_t)(b * NKV + kvh) * HD + d) * S_ + s] = (_Float16)acc[nt][v];
      }
    }
  }
}

// ---------------------------------------------------------------------------
// Flash attention: wave handles 16 query rows, key blocks of 32, causal.
// Writes Xattn [B,S,NH*HD] f16.
// ---------------------------------------------------------------------------
__global__ void __launch_bounds__(128)
flash_attn_kernel(const _Float16* __restrict__ Qh, const _Float16* __restrict__ Kh,
                  const _Float16* __restrict__ Vt, _Float16* __restrict__ Xattn) {
  __shared__ __align__(16) float plds[4][16 * 32];
  int lane = threadIdx.x & 31, wave = threadIdx.x >> 5;
  int bh = blockIdx.y;
  int b = bh / NH, h = bh % NH, kvh = h / REP;
  int q0 = (blockIdx.x * 4 + wave) * 16;
  int half = lane >> 4, nl = lane & 15;

  const _Float16* qptr = Qh + ((size_t)(b * NH + h) * S_ + q0) * HD;
  const _Float16* kptr = Kh + (size_t)(b * NKV + kvh) * S_ * HD;
  const _Float16* vptr = Vt + (size_t)(b * NKV + kvh) * HD * S_;
  float* pls = &plds[wave][0];

  v16h qf[4];
#pragma unroll
  for (int dc = 0; dc < 4; ++dc) qf[dc] = load_afrag(qptr + dc * 32, HD, lane);

  v8f o[8];
#pragma unroll
  for (int nt = 0; nt < 8; ++nt) o[nt] = v8f{};
  float mrow[8], lrow[8];
#pragma unroll
  for (int v = 0; v < 8; ++v) { mrow[v] = -1e30f; lrow[v] = 0.f; }

  for (int kb = 0; kb < q0 + 16; kb += 32) {
    // ---- S = Q * K^T over this key block (batch all K-fragment loads) ----
    v16h bk[8];
#pragma unroll
    for (int dc = 0; dc < 4; ++dc) {
      bk[dc]     = load_bfrag(kptr + (size_t)kb * HD + dc * 32, HD, lane);
      bk[4 + dc] = load_bfrag(kptr + (size_t)(kb + 16) * HD + dc * 32, HD, lane);
    }
    __builtin_amdgcn_sched_barrier(0);   // keep the 16 loads batched
    v8f s0 = v8f{}, s1 = v8f{};
#pragma unroll
    for (int dc = 0; dc < 4; ++dc) {
      s0 = wmma16(qf[dc], bk[dc], s0);
      s1 = wmma16(qf[dc], bk[4 + dc], s1);
    }
    // ---- online softmax (row m = v + 8*half in one 16-lane group; N = nl) ----
#pragma unroll
    for (int v = 0; v < 8; ++v) {
      int qrow = q0 + v + 8 * half;
      if (kb + nl > qrow)       s0[v] = -1e30f;   // causal mask
      if (kb + 16 + nl > qrow)  s1[v] = -1e30f;
      float mx = fmaxf(s0[v], s1[v]);
#pragma unroll
      for (int off = 8; off; off >>= 1) mx = fmaxf(mx, __shfl_xor(mx, off, 32));
      float mnew = fmaxf(mrow[v], mx);
      float sc = __expf(mrow[v] - mnew);
      mrow[v] = mnew;
      float p0 = __expf(s0[v] - mnew);
      float p1 = __expf(s1[v] - mnew);
      float rs = p0 + p1;
#pragma unroll
      for (int off = 8; off; off >>= 1) rs += __shfl_xor(rs, off, 32);
      lrow[v] = lrow[v] * sc + rs;
#pragma unroll
      for (int nt = 0; nt < 8; ++nt) o[nt][v] *= sc;
      int m = v + 8 * half;
      pls[m * 32 + nl] = p0;
      pls[m * 32 + 16 + nl] = p1;
    }
    // ---- batch V-fragment loads (overlap with LDS P bounce) ----
    v16h bv[8];
#pragma unroll
    for (int nt = 0; nt < 8; ++nt)
      bv[nt] = load_bfrag(vptr + (size_t)(nt * 16) * S_ + kb, S_, lane);
    __builtin_amdgcn_sched_barrier(0);   // pin V loads before the LDS reload
    asm volatile("s_wait_dscnt 0" ::: "memory");   // wave-local LDS RAW
    // reload P in A-fragment layout, convert to f16
    v16h pf;
    {
      const float* pr = pls + (lane & 15) * 32;
      int h8 = half * 8;
      v4f r0 = *(const v4f*)(pr + h8);
      v4f r1 = *(const v4f*)(pr + h8 + 4);
      v4f r2 = *(const v4f*)(pr + 16 + h8);
      v4f r3 = *(const v4f*)(pr + 16 + h8 + 4);
      _Float16* pe = (_Float16*)&pf;
#pragma unroll
      for (int e = 0; e < 4; ++e) {
        pe[e]      = (_Float16)r0[e];
        pe[4 + e]  = (_Float16)r1[e];
        pe[8 + e]  = (_Float16)r2[e];
        pe[12 + e] = (_Float16)r3[e];
      }
    }
    // ---- O += P * V ----
#pragma unroll
    for (int nt = 0; nt < 8; ++nt)
      o[nt] = wmma16(pf, bv[nt], o[nt]);
  }
  // normalize and store head slice
#pragma unroll
  for (int v = 0; v < 8; ++v) {
    float inv = 1.f / lrow[v];
    int s = q0 + v + 8 * half;
    size_t ob = ((size_t)b * S_ + s) * DM + (size_t)h * HD;
#pragma unroll
    for (int nt = 0; nt < 8; ++nt)
      Xattn[ob + nt * 16 + nl] = (_Float16)(o[nt][v] * inv);
  }
}

// ---------------------------------------------------------------------------
// Output projection: out = Xattn * Wo + bo (f32 out)
// ---------------------------------------------------------------------------
__global__ void __launch_bounds__(128)
oproj_kernel(const _Float16* __restrict__ Xattn, const _Float16* __restrict__ WoT,
             const float* __restrict__ bo, float* __restrict__ out) {
  int lane = threadIdx.x & 31, wave = threadIdx.x >> 5;
  int rowBase = (blockIdx.x * 4 + wave) * 16;
  int nBase = blockIdx.y * 128;
  v8f acc[8];
  wave_gemm_tile(Xattn + (size_t)rowBase * DM, DM, WoT + (size_t)nBase * DM, DM, DM, lane, acc);
  int half = lane >> 4, nl = lane & 15;
#pragma unroll
  for (int nt = 0; nt < 8; ++nt) {
    int n = nBase + nt * 16 + nl;
    float bb = bo[n];
#pragma unroll
    for (int v = 0; v < 8; ++v)
      out[(size_t)(rowBase + v + 8 * half) * DM + n] = acc[nt][v] + bb;
  }
}

// ---------------------------------------------------------------------------
extern "C" void kernel_launch(void* const* d_in, const int* in_sizes, int n_in,
                              void* d_out, int out_size, void* d_ws, size_t ws_size,
                              hipStream_t stream) {
  const float* x       = (const float*)d_in[0];
  const int*   seq_pos = (const int*)d_in[1];
  const float* wq      = (const float*)d_in[2];
  const float* bq      = (const float*)d_in[3];
  const float* wkv     = (const float*)d_in[4];
  const float* bkv     = (const float*)d_in[5];
  const float* wo      = (const float*)d_in[6];
  const float* bo      = (const float*)d_in[7];
  float* out = (float*)d_out;

  char* ws = (char*)d_ws;
  size_t off = 0;
  _Float16* Xh    = (_Float16*)(ws + off); off += (size_t)MROWS * DM * 2;          // 16 MB
  _Float16* WqT   = (_Float16*)(ws + off); off += (size_t)DM * DM * 2;             //  8 MB
  _Float16* WkvT  = (_Float16*)(ws + off); off += (size_t)(NKV * 2 * HD) * DM * 2; //  4 MB
  _Float16* WoT   = (_Float16*)(ws + off); off += (size_t)DM * DM * 2;             //  8 MB
  _Float16* Qh    = (_Float16*)(ws + off); off += (size_t)B_ * NH * S_ * HD * 2;   // 16 MB
  _Float16* Kh    = (_Float16*)(ws + off); off += (size_t)B_ * NKV * S_ * HD * 2;  //  4 MB
  _Float16* Vt    = (_Float16*)(ws + off); off += (size_t)B_ * NKV * HD * S_ * 2;  //  4 MB
  _Float16* Xattn = (_Float16*)(ws + off); off += (size_t)MROWS * DM * 2;          // 16 MB

  // pack
  cvt_x_kernel<<<(MROWS * DM) / 256, 256, 0, stream>>>(x, Xh, MROWS * DM);
  pack_wq_kernel<<<(DM * DM) / 256, 256, 0, stream>>>(wq, WqT);
  pack_wkv_kernel<<<((NKV * 2 * HD) * DM) / 256, 256, 0, stream>>>(wkv, WkvT);
  pack_wo_kernel<<<(DM * DM) / 256, 256, 0, stream>>>(wo, WoT);

  // projections (+RoPE)
  qproj_rope_kernel<<<dim3(MROWS / 64, DM / 128), 128, 0, stream>>>(Xh, WqT, bq, seq_pos, Qh);
  kvproj_kernel<<<dim3(MROWS / 64, (NKV * 2 * HD) / 128), 128, 0, stream>>>(Xh, WkvT, bkv, seq_pos, Kh, Vt);

  // attention
  flash_attn_kernel<<<dim3(S_ / 64, B_ * NH), 128, 0, stream>>>(Qh, Kh, Vt, Xattn);

  // output projection
  oproj_kernel<<<dim3(MROWS / 64, DM / 128), 128, 0, stream>>>(Xattn, WoT, bo, out);
}
